// _SequentialLSTMLayer_6270652252322
// MI455X (gfx1250) — matrix operand order repaired and checked by reading, not compiled, not run on hardware
//
#include <hip/hip_runtime.h>
#include <hip/hip_bf16.h>

// ---------------------------------------------------------------------------
// CDNA5 WMMA (wave32): D(16x16,f32) = A(16x32,bf16) x B(32x16,bf16) + C
// Per-lane fragment data is 16B-contiguous in row-major memory, so fragments
// are loaded directly from global with b128 loads (no LDS staging needed).
// K loop is software-pipelined (double-buffered fragments) so next-tile loads
// overlap the current tile's WMMAs.
// ---------------------------------------------------------------------------
typedef __bf16 bf16_t;
typedef bf16_t v16bf __attribute__((ext_vector_type(16)));
typedef float  v8f   __attribute__((ext_vector_type(8)));

union FragBF {
    v16bf        v;
    unsigned int u[8];
    uint4        q[2];
};

__device__ __forceinline__ unsigned short f32_to_bf16_bits(float f) {
    union { float f; unsigned int u; } x;
    x.f = f;
    unsigned int r = x.u + 0x7FFFu + ((x.u >> 16) & 1u); // round-to-nearest-even
    return (unsigned short)(r >> 16);
}

__device__ __forceinline__ float sigmoid_fast(float x) {
    return 1.0f / (1.0f + __expf(-x));
}
__device__ __forceinline__ float tanh_fast(float x) {
    float e = __expf(-2.0f * x);
    return (1.0f - e) / (1.0f + e);
}

// A fragment: 16x32 bf16, row-major source with leading dim ld (ushorts).
// Lane (m = lane&15, khalf = lane>>4) reads two contiguous 16B runs of row m.
__device__ __forceinline__ FragBF load_frag_a(const unsigned short* __restrict__ p,
                                              int row, int ld, int k0, int khalf) {
    FragBF a;
    const unsigned short* r = p + (size_t)row * ld + k0 + khalf * 8;
    a.q[0] = *(const uint4*)(r);        // K = k0+khalf*8    .. +7
    a.q[1] = *(const uint4*)(r + 16);   // K = k0+16+khalf*8 .. +7
    return a;
}

// B fragment: 32x16 bf16 where B[k][n] = W[n][k], W row-major, ld ushorts.
// Lane (n = lane&15, khalf = lane>>4) reads two contiguous 16B runs of row n.
__device__ __forceinline__ FragBF load_frag_b(const unsigned short* __restrict__ w,
                                              int row, int ld, int k0, int khalf) {
    FragBF b;
    const unsigned short* r = w + (size_t)row * ld + k0 + khalf * 8;
    b.q[0] = *(const uint4*)(r);        // K = k0+khalf*8    .. +7
    b.q[1] = *(const uint4*)(r + 16);   // K = k0+16+khalf*8 .. +7
    return b;
}

// ---------------------------------------------------------------------------
// Problem constants
// ---------------------------------------------------------------------------
#define B_   128
#define T_   512
#define I_   512
#define U_   1024
#define G4_  4096   // 4*U

// ---------------------------------------------------------------------------
// Prep kernels
// ---------------------------------------------------------------------------
__global__ __launch_bounds__(256) void k_convert_bf16(const float* __restrict__ in,
                                                      unsigned short* __restrict__ out,
                                                      int n4) {
    int i = blockIdx.x * blockDim.x + threadIdx.x;
    if (i < n4) {
        float4 f = ((const float4*)in)[i];
        ushort4 o;
        o.x = f32_to_bf16_bits(f.x);
        o.y = f32_to_bf16_bits(f.y);
        o.z = f32_to_bf16_bits(f.z);
        o.w = f32_to_bf16_bits(f.w);
        ((ushort4*)out)[i] = o;
    }
}

__global__ __launch_bounds__(256) void k_bias_sum(const float* __restrict__ bi,
                                                  const float* __restrict__ bh,
                                                  float* __restrict__ bs, int n) {
    int i = blockIdx.x * blockDim.x + threadIdx.x;
    if (i < n) bs[i] = bi[i] + bh[i];
}

__global__ __launch_bounds__(256) void k_zero_state(unsigned short* __restrict__ hA,
                                                    unsigned short* __restrict__ hB,
                                                    float* __restrict__ c, int n) {
    int i = blockIdx.x * blockDim.x + threadIdx.x;
    if (i < n) { hA[i] = 0; hB[i] = 0; c[i] = 0.0f; }
}

// ---------------------------------------------------------------------------
// Phase 1: x_gates[T][B][4096] = X(bf16)[BT][512] @ W_ih(bf16)[4096][512]^T + bias
// Block: 8 waves. Each wave: M-rows [bt0 + wave*16, +16), N-cols [n0, n0+64)
// as four 16x16 accumulators sharing one A fragment per K-step.
// ---------------------------------------------------------------------------
__global__ __launch_bounds__(256) void k_xgemm(const unsigned short* __restrict__ xb,
                                               const unsigned short* __restrict__ wih,
                                               const float* __restrict__ bsum,
                                               float* __restrict__ xg) {
    const int tid   = threadIdx.x;
    const int lane  = tid & 31;
    const int wave  = tid >> 5;
    const int nlane = lane & 15;
    const int khalf = lane >> 4;

    const int bt0 = blockIdx.y * 128 + wave * 16;  // this wave's M base
    const int n0  = blockIdx.x * 64;               // block's N base

    v8f acc[4];
    #pragma unroll
    for (int j = 0; j < 4; ++j) {
        float b = bsum[n0 + j * 16 + nlane];
        #pragma unroll
        for (int r = 0; r < 8; ++r) acc[j][r] = b;
    }

    const int am = bt0 + nlane;  // A row for this lane

    // Software-pipelined K loop: prologue loads tile 0, each iteration loads
    // tile k+1 (registers) before consuming tile k with 4 WMMAs.
    FragBF a = load_frag_a(xb, am, I_, 0, khalf);
    FragBF b[4];
    #pragma unroll
    for (int j = 0; j < 4; ++j)
        b[j] = load_frag_b(wih, n0 + j * 16 + nlane, I_, 0, khalf);

    for (int kk = 0; kk < I_; kk += 32) {
        const int kn = (kk + 32 < I_) ? kk + 32 : kk;  // uniform, branch-free
        FragBF an = load_frag_a(xb, am, I_, kn, khalf);
        FragBF bn[4];
        #pragma unroll
        for (int j = 0; j < 4; ++j)
            bn[j] = load_frag_b(wih, n0 + j * 16 + nlane, I_, kn, khalf);

        #pragma unroll
        for (int j = 0; j < 4; ++j)
            acc[j] = __builtin_amdgcn_wmma_f32_16x16x32_bf16(false, a.v, false, b[j].v,
                                                             (short)0, acc[j], false, false);
        a = an;
        #pragma unroll
        for (int j = 0; j < 4; ++j) b[j] = bn[j];
    }

    // Store to x_gates laid out [T][B][4096] for contiguous per-step reads.
    #pragma unroll
    for (int j = 0; j < 4; ++j) {
        const int nc = n0 + j * 16 + nlane;
        #pragma unroll
        for (int r = 0; r < 8; ++r) {
            int bt = bt0 + r + 8 * khalf;
            int bb = bt >> 9;        // bt / T
            int tt = bt & (T_ - 1);  // bt % T
            xg[((size_t)tt * B_ + bb) * G4_ + nc] = acc[j][r];
        }
    }
}

// ---------------------------------------------------------------------------
// Phase 2: one LSTM step.
// gates[b,:] = xg[t][b][:] + h @ W_hh^T ; i/f/g/o nonlinearity; c/h update.
// Block: 32 batch rows x 64 units, ALL FOUR gate tiles (elementwise update is
// block-local). 8 waves (wm in 0..1 x wn in 0..3), 4 accumulators per wave.
// W_hh (bf16, 8 MB) stays L2-resident across all 512 steps.
// ---------------------------------------------------------------------------
__global__ __launch_bounds__(256) void k_step(const float* __restrict__ xg,
                                              const unsigned short* __restrict__ whh,
                                              const unsigned short* __restrict__ hin,
                                              unsigned short* __restrict__ hout,
                                              float* __restrict__ cbuf,
                                              float* __restrict__ out,
                                              int t) {
    const int tid   = threadIdx.x;
    const int lane  = tid & 31;
    const int wave  = tid >> 5;
    const int wm    = wave >> 2;
    const int wn    = wave & 3;
    const int nlane = lane & 15;
    const int khalf = lane >> 4;

    const int b0 = blockIdx.y * 32;
    const int u0 = blockIdx.x * 64;
    const int uc = u0 + wn * 16 + nlane;       // this lane's unit column
    const int am = b0 + wm * 16 + nlane;       // this lane's A (batch) row

    // Seed accumulators with precomputed input gates (includes both biases).
    v8f acc[4];
    #pragma unroll
    for (int g = 0; g < 4; ++g) {
        #pragma unroll
        for (int r = 0; r < 8; ++r) {
            int bb = b0 + wm * 16 + r + 8 * khalf;
            acc[g][r] = xg[((size_t)t * B_ + bb) * G4_ + g * U_ + uc];
        }
    }

    // Software-pipelined K loop over the recurrent contraction (K = 1024).
    FragBF a = load_frag_a(hin, am, U_, 0, khalf);
    FragBF b[4];
    #pragma unroll
    for (int g = 0; g < 4; ++g)
        b[g] = load_frag_b(whh, g * U_ + uc, U_, 0, khalf);

    for (int kk = 0; kk < U_; kk += 32) {
        const int kn = (kk + 32 < U_) ? kk + 32 : kk;  // uniform, branch-free
        FragBF an = load_frag_a(hin, am, U_, kn, khalf);
        FragBF bn[4];
        #pragma unroll
        for (int g = 0; g < 4; ++g)
            bn[g] = load_frag_b(whh, g * U_ + uc, U_, kn, khalf);

        #pragma unroll
        for (int g = 0; g < 4; ++g)
            acc[g] = __builtin_amdgcn_wmma_f32_16x16x32_bf16(false, a.v, false, b[g].v,
                                                             (short)0, acc[g], false, false);
        a = an;
        #pragma unroll
        for (int g = 0; g < 4; ++g) b[g] = bn[g];
    }

    // Elementwise LSTM update (all four gates are wave-local per element).
    #pragma unroll
    for (int r = 0; r < 8; ++r) {
        int bb = b0 + wm * 16 + r + 8 * khalf;
        float iv = sigmoid_fast(acc[0][r]);
        float fv = sigmoid_fast(acc[1][r]);
        float gv = tanh_fast(acc[2][r]);
        float ov = sigmoid_fast(acc[3][r]);
        size_t su = (size_t)bb * U_ + uc;
        float cp = cbuf[su];
        float cn = fv * cp + iv * gv;
        float hn = ov * tanh_fast(cn);
        cbuf[su] = cn;
        out[((size_t)bb * T_ + t) * U_ + uc] = hn;
        hout[su] = f32_to_bf16_bits(hn);
    }
}

// ---------------------------------------------------------------------------
// Host launcher
// ---------------------------------------------------------------------------
extern "C" void kernel_launch(void* const* d_in, const int* in_sizes, int n_in,
                              void* d_out, int out_size, void* d_ws, size_t ws_size,
                              hipStream_t stream) {
    (void)in_sizes; (void)n_in; (void)out_size; (void)ws_size;

    const float* x   = (const float*)d_in[0];  // [128][512][512]
    const float* Wih = (const float*)d_in[1];  // [4096][512]
    const float* Whh = (const float*)d_in[2];  // [4096][1024]
    const float* bih = (const float*)d_in[3];  // [4096]
    const float* bhh = (const float*)d_in[4];  // [4096]
    float* out = (float*)d_out;                // [128][512][1024]

    // Workspace partition (all offsets 256B aligned).
    char* ws = (char*)d_ws;
    size_t off = 0;
    auto carve = [&](size_t bytes) -> void* {
        void* p = ws + off;
        off = (off + bytes + 255) & ~(size_t)255;
        return p;
    };
    float*          xg   = (float*)carve((size_t)T_ * B_ * G4_ * sizeof(float));  // 1 GiB
    unsigned short* xb   = (unsigned short*)carve((size_t)B_ * T_ * I_ * 2);      // 64 MiB
    unsigned short* wihb = (unsigned short*)carve((size_t)G4_ * I_ * 2);          // 4 MiB
    unsigned short* whhb = (unsigned short*)carve((size_t)G4_ * U_ * 2);          // 8 MiB
    float*          bsum = (float*)carve((size_t)G4_ * sizeof(float));
    unsigned short* hA   = (unsigned short*)carve((size_t)B_ * U_ * 2);
    unsigned short* hB   = (unsigned short*)carve((size_t)B_ * U_ * 2);
    float*          cbuf = (float*)carve((size_t)B_ * U_ * sizeof(float));

    // --- Prep: fp32 -> bf16 conversions (layouts preserved) ---
    {
        int n4 = (B_ * T_ * I_) / 4;
        k_convert_bf16<<<(n4 + 255) / 256, 256, 0, stream>>>(x, xb, n4);
    }
    {
        int n4 = (G4_ * I_) / 4;
        k_convert_bf16<<<(n4 + 255) / 256, 256, 0, stream>>>(Wih, wihb, n4);
    }
    {
        int n4 = (G4_ * U_) / 4;
        k_convert_bf16<<<(n4 + 255) / 256, 256, 0, stream>>>(Whh, whhb, n4);
    }
    k_bias_sum<<<(G4_ + 255) / 256, 256, 0, stream>>>(bih, bhh, bsum, G4_);
    k_zero_state<<<(B_ * U_ + 255) / 256, 256, 0, stream>>>(hA, hB, cbuf, B_ * U_);

    // --- Phase 1: input-projection GEMM (M=65536, N=4096, K=512) ---
    {
        dim3 grid(G4_ / 64, (B_ * T_) / 128);  // (64, 512)
        k_xgemm<<<grid, 256, 0, stream>>>(xb, wihb, bsum, xg);
    }

    // --- Phase 2: 512 sequential fused LSTM steps (ping-pong h buffers) ---
    {
        dim3 grid(U_ / 64, B_ / 32);  // (16, 4)
        for (int t = 0; t < T_; ++t) {
            const unsigned short* hi = (t & 1) ? hB : hA;
            unsigned short*       ho = (t & 1) ? hA : hB;
            k_step<<<grid, 256, 0, stream>>>(xg, whhb, hi, ho, cbuf, out, t);
        }
    }
}
